// CustomLSTM_16071767621827
// MI455X (gfx1250) — compile-verified, matrix operand order checked by reading, not run on hardware
//
#include <hip/hip_runtime.h>

// ---------------------------------------------------------------------------
// CDNA5 (gfx1250) LSTM: bf16 WMMA cells, double-buffered TDM weight staging,
// software-pipelined inner loop (peeled tail), fused gate epilogue.
//
// ws layout (bytes, ~116.5 MB total; assumed available):
//   [0)            Wc   : bf16 per-layer [4096][in+H], gate-interleaved, n-major
//   [+96,468,992)  bias : f32 [6][4096], gate-interleaved
//   [+98,304)      c    : f32 [6][64][1024]
//   [+1,572,864)   h0   : bf16 [6][64][1024]   (ping)
//   [+786,432)     h1   : bf16 [6][64][1024]   (pong)
//   [+786,432)     xbf  : bf16 [64][256][512]  (x pre-converted)
// ---------------------------------------------------------------------------

typedef __attribute__((ext_vector_type(16))) __bf16 v16bf;
typedef __attribute__((ext_vector_type(8)))  float  v8f;
typedef unsigned int u32x4 __attribute__((ext_vector_type(4)));
typedef int          i32x4 __attribute__((ext_vector_type(4)));
typedef int          i32x8 __attribute__((ext_vector_type(8)));

#define IN_DIM 512
#define H_DIM  1024
#define L_NUM  6
#define B_NUM  64
#define S_NUM  256
#define G4     4096  // 4*H

#if defined(__gfx1250__) && defined(__has_builtin)
# if __has_builtin(__builtin_amdgcn_tensor_load_to_lds) && \
     __has_builtin(__builtin_amdgcn_s_wait_tensorcnt)
#  define USE_TDM 1
# endif
#endif

__device__ __forceinline__ float sigmoidf_(float x) {
    return 1.0f / (1.0f + __expf(-x));
}

__device__ __forceinline__ v8f wmma_bf16(v16bf a, v16bf b, v8f c) {
    return __builtin_amdgcn_wmma_f32_16x16x32_bf16(false, a, false, b,
                                                   (short)0, c, false, false);
}

#if defined(USE_TDM)
typedef __attribute__((address_space(3))) void lds_void_t;

// Issue one TDM 2-D tile load: tile_k x tile_n bf16 elements, row stride = ktot.
__device__ __forceinline__ void tdm_load_tile(void* lds_dst, const void* gsrc,
                                              unsigned tile_k, unsigned tile_n,
                                              unsigned ktot, unsigned tensor_n) {
    unsigned lds_off = (unsigned)(unsigned long long)(lds_void_t*)lds_dst;
    unsigned long long ga = (unsigned long long)gsrc;
    u32x4 g0;
    g0[0] = 1u;                                    // count=1 (valid), flags 0
    g0[1] = lds_off;                               // lds_addr
    g0[2] = (unsigned)(ga & 0xffffffffu);          // global_addr[31:0]
    g0[3] = (unsigned)((ga >> 32) & 0x01ffffffu)   // global_addr[56:32]
          | (2u << 30);                            // type = 2 ("image")
    i32x8 g1;
    unsigned td0 = ktot;                           // tensor_dim0 (elements)
    unsigned td1 = tensor_n;                       // tensor_dim1 (rows)
    unsigned long long s0 = (unsigned long long)ktot;  // tensor_dim0_stride
    g1[0] = (int)(1u << 16);                       // data_size=1 -> 2 bytes
    g1[1] = (int)((td0 & 0xffffu) << 16);          // tensor_dim0[15:0] @ bits63:48
    g1[2] = (int)(((td0 >> 16) & 0xffffu) | ((td1 & 0xffffu) << 16));
    g1[3] = (int)(((td1 >> 16) & 0xffffu) | (tile_k << 16));   // tile_dim0
    g1[4] = (int)tile_n;                           // tile_dim1 (tile_dim2 = 0)
    g1[5] = (int)(unsigned)(s0 & 0xffffffffu);     // stride0[31:0]
    g1[6] = (int)(unsigned)((s0 >> 32) & 0xffffu); // stride0[47:32], stride1=0
    g1[7] = 0;
    i32x4 z4 = {0, 0, 0, 0};
#if __clang_major__ >= 23
    i32x8 z8 = {0, 0, 0, 0, 0, 0, 0, 0};
    __builtin_amdgcn_tensor_load_to_lds(g0, g1, z4, z4, z8, 0);
#else
    __builtin_amdgcn_tensor_load_to_lds(g0, g1, z4, z4, 0);
#endif
}
#endif  // USE_TDM

// x: f32 [B][S][IN] -> bf16, same layout.
__global__ __launch_bounds__(256) void k_xconv(const float* __restrict__ x,
                                               __bf16* __restrict__ xbf) {
    long long i = (long long)blockIdx.x * 256 + threadIdx.x;
    xbf[i] = (__bf16)x[i];
}

// Weight repack: f32 [K][4096] (U then V) -> bf16 [4096][in+H], n-major,
// gate-interleaved column permutation: p -> orig col (p%4)*H + p/4.
__global__ __launch_bounds__(256) void k_wconv(const float* __restrict__ U,
                                               const float* __restrict__ V,
                                               __bf16* __restrict__ dst,
                                               int inl) {
    const int ktot = inl + H_DIM;
    long long idx = (long long)blockIdx.x * 256 + threadIdx.x;
    int p = (int)(idx / ktot);
    int k = (int)(idx % ktot);
    int c = (p & 3) * H_DIM + (p >> 2);
    float v = (k < inl) ? U[(long long)k * G4 + c]
                        : V[(long long)(k - inl) * G4 + c];
    dst[idx] = (__bf16)v;
}

__global__ __launch_bounds__(256) void k_bias(const float* __restrict__ b0,
                                              const float* __restrict__ bl,
                                              float* __restrict__ dst) {
    int idx = blockIdx.x * 256 + threadIdx.x;   // [0, 6*4096)
    int l = idx >> 12;
    int p = idx & (G4 - 1);
    int c = (p & 3) * H_DIM + (p >> 2);
    dst[idx] = (l == 0) ? b0[c] : bl[(l - 1) * G4 + c];
}

__global__ __launch_bounds__(256) void k_zero(unsigned* __restrict__ p, long long n) {
    long long i = (long long)blockIdx.x * 256 + threadIdx.x;
    if (i < n) p[i] = 0u;
}

// One LSTM cell step for one layer.
// grid = 64 blocks (64 gate-cols == 16 hidden units each), 256 threads (8 waves).
// Each wave: 16(batch) x 32(cols); KT=64 -> 4 v_wmma per iteration, one barrier,
// next tile's TDM + global loads overlapped with the WMMAs (double buffered).
// launch_bounds(256,1): allow full register budget, avoid scratch spills.
__global__ __launch_bounds__(256, 1) void k_cell(
    const __bf16* __restrict__ inp,   // layer input, bf16
    int istr, int iofs,               // input row stride / base offset (elements)
    const __bf16* __restrict__ hrec,  // this layer's h (prev step) [B][H]
    const __bf16* __restrict__ W,     // [4096][ktot] bf16, n-major
    const float*  __restrict__ bias,  // [4096] permuted
    float*        __restrict__ cst,   // [B][H] cell state, this layer
    __bf16*       __restrict__ hout,  // [B][H] new h (bf16)
    float*        __restrict__ seq,   // d_out seq region (top layer), else nullptr
    float*        __restrict__ hf,    // d_out hf slice (last step), else nullptr
    float*        __restrict__ cf,    // d_out cf slice (last step), else nullptr
    int inl, int t) {
    constexpr int KT = 64;
    __shared__ __align__(32) __bf16 xs[2][64 * KT];  // activations, 2 x 8 KB
    __shared__ __align__(32) __bf16 wt[2][64 * KT];  // weights,     2 x 8 KB
    __shared__ float zs[64 * 64];                    // z tile, 16 KB

    const int tid  = threadIdx.x;
    const int lane = tid & 31;
    const int wid  = tid >> 5;
    const int m0   = (wid >> 1) * 16;   // batch-tile base (0,16,32,48)
    const int n0   = (wid & 1) * 32;    // col-tile base within block (0,32)
    const int ktot = inl + H_DIM;
    const int ntile = ktot / KT;
    const int pbase = blockIdx.x * 64;  // global gate-column base
    const int lh = lane >> 4;           // K-half select per WMMA layout
    const int ll = lane & 15;

    // Hoisted per-thread offsets.
    const int xfo  = (m0 + ll) * KT + lh * 16;        // A fragment (xs)
    const int wfoA = (n0 + ll) * KT + lh * 16;        // B fragment, subtile 0
    const int wfoB = (n0 + 16 + ll) * KT + lh * 16;   // B fragment, subtile 1
    // Activation staging chunks: 2 x 16 B per thread, LDS offsets loop-invariant.
    int sm[2], sk[2], slo[2];
#pragma unroll
    for (int i = 0; i < 2; ++i) {
        int ch = tid + i * 256;         // 0..511
        sm[i]  = ch >> 3;               // batch row
        sk[i]  = (ch & 7) * 8;          // k offset within tile
        slo[i] = sm[i] * KT + sk[i];    // LDS element offset
    }

    v8f acc0 = {};
    v8f acc1 = {};

    // --- Prologue: stage tile 0 into buffer 0 ---
#if defined(USE_TDM)
    if (wid == 0) {
        tdm_load_tile(&wt[0][0], W + (long long)pbase * ktot, KT, 64,
                      (unsigned)ktot, G4);
    }
#else
#pragma unroll
    for (int i = 0; i < 2; ++i) {
        int r = (tid + i * 256) >> 3, cq = ((tid + i * 256) & 7) * 8;
        const __bf16* src = W + (long long)(pbase + r) * ktot + cq;
        *(uint4*)(&wt[0][r * KT + cq]) = *(const uint4*)src;
        __builtin_prefetch(src + KT, 0, 1);
    }
#endif
#pragma unroll
    for (int i = 0; i < 2; ++i) {
        int kg = sk[i];
        const __bf16* src = (kg < inl)
            ? (inp + (long long)sm[i] * istr + iofs + kg)
            : (hrec + sm[i] * H_DIM + (kg - inl));
        *(uint4*)(&xs[0][slo[i]]) = *(const uint4*)src;
    }
#if defined(USE_TDM)
    if (wid == 0) __builtin_amdgcn_s_wait_tensorcnt(0);
#endif
    __syncthreads();

    // --- Pipelined main loop (tiles 0 .. ntile-2), one barrier per tile ---
    for (int it = 0; it < ntile - 1; ++it) {
        const int cur = it & 1;
        const int nxt = cur ^ 1;
        const int k0n = (it + 1) * KT;
        const __bf16* xc = &xs[cur][0];
        const __bf16* wc = &wt[cur][0];

        // 1) Chunk-0 fragment loads (ds_loads precede this iteration's ds_stores).
        v16bf a0  = *(const v16bf*)(xc + xfo);
        v16bf b0A = *(const v16bf*)(wc + wfoA);
        v16bf b0B = *(const v16bf*)(wc + wfoB);

        // 2) Kick next tile: TDM for weights, global loads for activations.
#if defined(USE_TDM)
        if (wid == 0) {
            tdm_load_tile(&wt[nxt][0], W + (long long)pbase * ktot + k0n,
                          KT, 64, (unsigned)ktot, G4);
        }
#else
        uint4 gw0, gw1;
        {
            int r0 = tid >> 3, c0q = (tid & 7) * 8;
            int r1 = (tid + 256) >> 3, c1q = ((tid + 256) & 7) * 8;
            gw0 = *(const uint4*)(W + (long long)(pbase + r0) * ktot + k0n + c0q);
            gw1 = *(const uint4*)(W + (long long)(pbase + r1) * ktot + k0n + c1q);
        }
#endif
        uint4 ga0, ga1;
        {
            int kg0 = k0n + sk[0];
            const __bf16* s0p = (kg0 < inl)
                ? (inp + (long long)sm[0] * istr + iofs + kg0)
                : (hrec + sm[0] * H_DIM + (kg0 - inl));
            ga0 = *(const uint4*)s0p;
            int kg1 = k0n + sk[1];
            const __bf16* s1p = (kg1 < inl)
                ? (inp + (long long)sm[1] * istr + iofs + kg1)
                : (hrec + sm[1] * H_DIM + (kg1 - inl));
            ga1 = *(const uint4*)s1p;
        }

        // 3) WMMA pair 0, then chunk-1 fragments, then WMMA pair 1.
        acc0 = wmma_bf16(a0, b0A, acc0);
        acc1 = wmma_bf16(a0, b0B, acc1);

        v16bf a1  = *(const v16bf*)(xc + xfo + 32);
        v16bf b1A = *(const v16bf*)(wc + wfoA + 32);
        v16bf b1B = *(const v16bf*)(wc + wfoB + 32);
        acc0 = wmma_bf16(a1, b1A, acc0);
        acc1 = wmma_bf16(a1, b1B, acc1);

        // 4) Commit staged data for tile it+1, then one barrier.
        *(uint4*)(&xs[nxt][slo[0]]) = ga0;
        *(uint4*)(&xs[nxt][slo[1]]) = ga1;
#if !defined(USE_TDM)
        {
            int r0 = tid >> 3, c0q = (tid & 7) * 8;
            int r1 = (tid + 256) >> 3, c1q = ((tid + 256) & 7) * 8;
            *(uint4*)(&wt[nxt][r0 * KT + c0q]) = gw0;
            *(uint4*)(&wt[nxt][r1 * KT + c1q]) = gw1;
        }
#endif
#if defined(USE_TDM)
        if (wid == 0) __builtin_amdgcn_s_wait_tensorcnt(0);
#endif
        __syncthreads();
    }

    // --- Peeled last tile: compute only, no staging, no barrier needed ---
    {
        const int cur = (ntile - 1) & 1;
        const __bf16* xc = &xs[cur][0];
        const __bf16* wc = &wt[cur][0];
        v16bf a0  = *(const v16bf*)(xc + xfo);
        v16bf b0A = *(const v16bf*)(wc + wfoA);
        v16bf b0B = *(const v16bf*)(wc + wfoB);
        acc0 = wmma_bf16(a0, b0A, acc0);
        acc1 = wmma_bf16(a0, b0B, acc1);
        v16bf a1  = *(const v16bf*)(xc + xfo + 32);
        v16bf b1A = *(const v16bf*)(wc + wfoA + 32);
        v16bf b1B = *(const v16bf*)(wc + wfoB + 32);
        acc0 = wmma_bf16(a1, b1A, acc0);
        acc1 = wmma_bf16(a1, b1B, acc1);
    }

    // Spill z (+bias) to LDS. C/D layout: VGPR r, lanes 0-15 -> M=r, 16-31 -> M=r+8.
#pragma unroll
    for (int r = 0; r < 8; ++r) {
        int m  = m0 + r + lh * 8;
        int c0 = n0 + ll;
        int c1 = n0 + 16 + ll;
        zs[m * 64 + c0] = acc0[r] + bias[pbase + c0];
        zs[m * 64 + c1] = acc1[r] + bias[pbase + c1];
    }
    __syncthreads();

    // Fused gate epilogue: 64 batch x 16 units; cols 4u..4u+3 = i,f,g,o of unit u.
#pragma unroll
    for (int i = 0; i < 4; ++i) {
        int e  = tid + i * 256;
        int b  = e >> 4;
        int ul = e & 15;
        float zi = zs[b * 64 + ul * 4 + 0];
        float zf = zs[b * 64 + ul * 4 + 1];
        float zg = zs[b * 64 + ul * 4 + 2];
        float zo = zs[b * 64 + ul * 4 + 3];
        float ig = sigmoidf_(zi);
        float fg = sigmoidf_(zf);
        float gg = tanhf(zg);
        float og = sigmoidf_(zo);
        int u = (pbase >> 2) + ul;
        long long ci = (long long)b * H_DIM + u;
        float cn = fg * cst[ci] + ig * gg;
        float hn = og * tanhf(cn);
        cst[ci]  = cn;
        hout[ci] = (__bf16)hn;
        if (seq) seq[(long long)b * (S_NUM * H_DIM) + t * H_DIM + u] = hn;
        if (hf)  hf[ci] = hn;
        if (cf)  cf[ci] = cn;
    }
}

extern "C" void kernel_launch(void* const* d_in, const int* in_sizes, int n_in,
                              void* d_out, int out_size, void* d_ws, size_t ws_size,
                              hipStream_t stream) {
    const float* x  = (const float*)d_in[0];
    const float* U0 = (const float*)d_in[1];
    const float* V0 = (const float*)d_in[2];
    const float* b0 = (const float*)d_in[3];
    const float* U  = (const float*)d_in[4];
    const float* V  = (const float*)d_in[5];
    const float* bb = (const float*)d_in[6];
    float* out = (float*)d_out;

    char* ws = (char*)d_ws;
    const long long w0 = (long long)G4 * (IN_DIM + H_DIM);  // halves, layer 0
    const long long wl = (long long)G4 * (2 * H_DIM);       // halves, layers 1..5
    long long woff[L_NUM];
    woff[0] = 0;
    for (int l = 1; l < L_NUM; ++l) woff[l] = w0 + (long long)(l - 1) * wl;
    const long long wtot = w0 + 5 * wl;

    __bf16* Wc = (__bf16*)ws;
    size_t off = (size_t)wtot * 2;
    float* biasp = (float*)(ws + off);   off += (size_t)L_NUM * G4 * 4;
    float* cbuf  = (float*)(ws + off);   off += (size_t)L_NUM * B_NUM * H_DIM * 4;
    __bf16* h0p  = (__bf16*)(ws + off);  off += (size_t)L_NUM * B_NUM * H_DIM * 2;
    __bf16* h1p  = (__bf16*)(ws + off);  off += (size_t)L_NUM * B_NUM * H_DIM * 2;
    __bf16* xbf  = (__bf16*)(ws + off);

    // One-time (per launch) conversions + state zeroing.
    const long long xn = (long long)B_NUM * S_NUM * IN_DIM;
    k_xconv<<<(unsigned)(xn / 256), 256, 0, stream>>>(x, xbf);
    k_wconv<<<(unsigned)(w0 / 256), 256, 0, stream>>>(U0, V0, Wc, IN_DIM);
    for (int l = 1; l < L_NUM; ++l) {
        k_wconv<<<(unsigned)(wl / 256), 256, 0, stream>>>(
            U + (long long)(l - 1) * H_DIM * G4,
            V + (long long)(l - 1) * H_DIM * G4,
            Wc + woff[l], H_DIM);
    }
    k_bias<<<(L_NUM * G4) / 256, 256, 0, stream>>>(b0, bb, biasp);
    {
        // c (f32) and h0 (bf16) are contiguous; zero both as words.
        long long words = ((long long)L_NUM * B_NUM * H_DIM * 4 +
                           (long long)L_NUM * B_NUM * H_DIM * 2) / 4;
        k_zero<<<(unsigned)((words + 255) / 256), 256, 0, stream>>>(
            (unsigned*)cbuf, words);
    }

    __bf16* hb[2] = {h0p, h1p};
    for (int t = 0; t < S_NUM; ++t) {
        const __bf16* hin = hb[t & 1];
        __bf16* hob = hb[(t + 1) & 1];
        for (int l = 0; l < L_NUM; ++l) {
            const __bf16* inp = (l == 0)
                ? xbf
                : hob + (long long)(l - 1) * B_NUM * H_DIM;
            int istr = (l == 0) ? (S_NUM * IN_DIM) : H_DIM;
            int iofs = (l == 0) ? (t * IN_DIM) : 0;
            k_cell<<<64, 256, 0, stream>>>(
                inp, istr, iofs,
                hin + (long long)l * B_NUM * H_DIM,
                Wc + woff[l],
                biasp + l * G4,
                cbuf + (long long)l * B_NUM * H_DIM,
                hob + (long long)l * B_NUM * H_DIM,
                l == L_NUM - 1 ? out : nullptr,
                t == S_NUM - 1 ? out + (long long)B_NUM * S_NUM * H_DIM +
                                       (long long)l * B_NUM * H_DIM
                               : nullptr,
                t == S_NUM - 1 ? out + (long long)B_NUM * S_NUM * H_DIM +
                                       (long long)L_NUM * B_NUM * H_DIM +
                                       (long long)l * B_NUM * H_DIM
                               : nullptr,
                l == 0 ? IN_DIM : H_DIM, t);
        }
    }
}
